// MarketTransformer_64579128263108
// MI455X (gfx1250) — matrix operand order repaired and checked by reading, not compiled, not run on hardware
//
#include <hip/hip_runtime.h>
#include <hip/hip_bf16.h>
#include <math.h>

#define NND 8192
#define HD 128
#define NHEADS 8
#define DHD 16
#define NEDGE 524288
#define FFD 2048

typedef __attribute__((ext_vector_type(16))) _Float16 v16h;
typedef __attribute__((ext_vector_type(4)))  _Float16 v4h;
typedef __attribute__((ext_vector_type(2)))  _Float16 v2h;
typedef __attribute__((ext_vector_type(8)))  float    v8f;

// ---------------- embed + sinusoidal positional encoding ----------------
__global__ void k_embed_pe(const float* __restrict__ x, const float* __restrict__ w,
                           const float* __restrict__ b, float* __restrict__ h) {
  int idx = blockIdx.x * blockDim.x + threadIdx.x;
  if (idx >= NND * HD) return;
  int n = idx / HD, c = idx % HD;
  float acc = b[c];
#pragma unroll
  for (int k = 0; k < 4; ++k) acc += x[n * 4 + k] * w[c * 4 + k];
  int j = c & ~1;                       // reference uses log(100000)
  float div = expf(-(float)j * (logf(100000.0f) / (float)HD));
  float arg = (float)n * div;
  acc += (c & 1) ? cosf(arg) : sinf(arg);
  h[idx] = acc;
}

// ------- block-cooperative WMMA GEMM: C = act(A[M,K] x W[Nout,K]^T + bias) -------
// block = 8 waves = one 16(M) x 128(N) strip; shared A tile, per-wave B tile.
// A tile stored [m][k] (32 halves/row); B tile stored [n][k] (32 halves/row) so both
// fragment gathers fold into contiguous ds_load_b128 pairs.
__global__ __launch_bounds__(256) void k_gemm_wmma(
    const float* __restrict__ A, const float* __restrict__ W,
    const float* __restrict__ bias, float* __restrict__ C,
    int M, int K, int Nout, int relu) {
  __shared__ __align__(16) _Float16 lsA[16 * 32];      // shared by all 8 waves
  __shared__ __align__(16) _Float16 lsB[8][16 * 32];   // per-wave weight tile [n][k]
  const int tid = threadIdx.x;
  const int wave = tid >> 5, lane = tid & 31;
  const int lo = lane & 15, hi = lane >> 4;
  const int tilesM = M >> 4;
  const int m0 = (blockIdx.x % tilesM) << 4;       // M fastest -> weight reuse in L2
  const int n0 = (blockIdx.x / tilesM) << 7;       // 128-wide strip
  const int nw = n0 + (wave << 4);

  v8f acc = {};
  for (int k0 = 0; k0 < K; k0 += 32) {
    __syncthreads();
    // stage shared A tile 16x32: 256 threads x float2 -> packed b32 DS store
    {
      int row = tid >> 4, pr = tid & 15;
      const float2 av = *(const float2*)(A + (size_t)(m0 + row) * K + k0 + pr * 2);
      v2h ap = {(_Float16)av.x, (_Float16)av.y};
      *(v2h*)(&lsA[row * 32 + pr * 2]) = ap;
    }
    // stage per-wave B tile [n][k]: 32 lanes x 4 float4 (b128) -> packed b64 DS store
#pragma unroll
    for (int j = 0; j < 4; ++j) {
      int idx = j * 32 + lane;
      int n = idx >> 3, kq = idx & 7;
      const float4 wv = *(const float4*)(W + (size_t)(nw + n) * K + k0 + kq * 4);
      v4h wp = {(_Float16)wv.x, (_Float16)wv.y, (_Float16)wv.z, (_Float16)wv.w};
      *(v4h*)(&lsB[wave][n * 32 + kq * 4]) = wp;
    }
    // prefetch next K slice (lowers to global_prefetch_b8)
    if (k0 + 32 < K) {
      if (tid < 16)
        __builtin_prefetch(A + (size_t)(m0 + tid) * K + k0 + 32, 0, 0);
      if (lane < 16)
        __builtin_prefetch(W + (size_t)(nw + lane) * K + k0 + 32, 0, 0);
    }
    __syncthreads();
    // fragment gathers: A and B = one contiguous 16-half run each
    v16h af, bf;
#pragma unroll
    for (int e = 0; e < 16; ++e) {
      int v = e >> 1, p = e & 1;
      int ka = hi * 16 + e;
      af[e] = lsA[lo * 32 + ka];
      bf[e] = lsB[wave][lo * 32 + hi * 16 + e];
    }
    acc = __builtin_amdgcn_wmma_f32_16x16x32_f16(false, af, false, bf,
                                                 (short)0, acc, false, false);
  }
#pragma unroll
  for (int r = 0; r < 8; ++r) {
    int row = m0 + r + 8 * hi, col = nw + lo;
    float v = acc[r] + bias[col];
    if (relu) v = fmaxf(v, 0.0f);
    C[(size_t)row * Nout + col] = v;
  }
}

// ---------------- flash attention: qkv[N][384] -> O[N][128] ----------------
// one (16-query tile, head) per wave; online softmax; dh=16 padded to K=32.
__global__ __launch_bounds__(256) void k_flash_attn(
    const float* __restrict__ qkv, float* __restrict__ O) {
  __shared__ __align__(16) float lsP[8][256];
  __shared__ __align__(16) float lsV[8][256];
  const int wave = threadIdx.x >> 5, lane = threadIdx.x & 31;
  const int lo = lane & 15, hi = lane >> 4;
  const int task = blockIdx.x * 8 + wave;
  const int head = task % NHEADS, qt = task / NHEADS;
  const int m0 = qt << 4;

  // Q fragment (A layout, K padded 16->32), pre-scaled by 1/sqrt(dh)=0.25
  v16h qf;
#pragma unroll
  for (int e = 0; e < 16; ++e) {
    int v = e >> 1, p = e & 1;
    int k = hi * 16 + e;
    qf[e] = (k < DHD)
        ? (_Float16)(0.25f * qkv[(size_t)(m0 + lo) * 384 + head * DHD + k])
        : (_Float16)0.0f;
  }

  float mrow[8], lrow[8];
  v8f oacc = {};
#pragma unroll
  for (int r = 0; r < 8; ++r) { mrow[r] = -3.0e38f; lrow[r] = 0.0f; }

  for (int kt = 0; kt < NND / 16; ++kt) {
    __syncthreads();   // previous iteration's lsP/lsV readers are done
    // stage V tile 16x16 via coalesced b128 rows: 32 lanes x 2 float4
#pragma unroll
    for (int j = 0; j < 2; ++j) {
      int idx = j * 32 + lane;
      int row = idx >> 2, q = idx & 3;
      const float4 vv = *(const float4*)(qkv + (size_t)(kt * 16 + row) * 384 +
                                         2 * HD + head * DHD + q * 4);
      *(float4*)(&lsV[wave][row * 16 + q * 4]) = vv;
    }
    // K^T fragment: B[k][n] = K[key=n][dim=k]; dims contiguous -> 4x b128
    v16h kf;
    if (hi == 0) {
      const float4* kp = (const float4*)(qkv + (size_t)(kt * 16 + lo) * 384 +
                                         HD + head * DHD);
      float4 a = kp[0], b = kp[1], c = kp[2], d = kp[3];
      kf[0] = (_Float16)a.x;  kf[1] = (_Float16)a.y;
      kf[2] = (_Float16)a.z;  kf[3] = (_Float16)a.w;
      kf[4] = (_Float16)b.x;  kf[5] = (_Float16)b.y;
      kf[6] = (_Float16)b.z;  kf[7] = (_Float16)b.w;
      kf[8] = (_Float16)c.x;  kf[9] = (_Float16)c.y;
      kf[10] = (_Float16)c.z; kf[11] = (_Float16)c.w;
      kf[12] = (_Float16)d.x; kf[13] = (_Float16)d.y;
      kf[14] = (_Float16)d.z; kf[15] = (_Float16)d.w;
    } else {
#pragma unroll
      for (int e = 0; e < 16; ++e) kf[e] = (_Float16)0.0f;
    }
    v8f s = {};
    s = __builtin_amdgcn_wmma_f32_16x16x32_f16(false, qf, false, kf,
                                               (short)0, s, false, false);
    // online softmax per row (reduce across 16-lane groups)
    float pr[8];
#pragma unroll
    for (int r = 0; r < 8; ++r) {
      float rowmax = s[r];
      for (int d = 1; d < 16; d <<= 1)
        rowmax = fmaxf(rowmax, __shfl_xor(rowmax, d, 32));
      float mnew = fmaxf(mrow[r], rowmax);
      float corr = expf(mrow[r] - mnew);
      float pe = expf(s[r] - mnew);
      float rs = pe;
      for (int d = 1; d < 16; d <<= 1) rs += __shfl_xor(rs, d, 32);
      lrow[r] = lrow[r] * corr + rs;
      mrow[r] = mnew;
      oacc[r] *= corr;
      pr[r] = pe;
    }
#pragma unroll
    for (int r = 0; r < 8; ++r) lsP[wave][(r + 8 * hi) * 16 + lo] = pr[r];
    __syncthreads();   // lsP/lsV writes visible
    // P as A fragment (K = 16 keys padded to 32); valid half is contiguous in LDS
    v16h pf;
#pragma unroll
    for (int e = 0; e < 16; ++e) {
      int v = e >> 1, p = e & 1;
      int k = hi * 16 + e;
      pf[e] = (k < 16) ? (_Float16)lsP[wave][lo * 16 + k] : (_Float16)0.0f;
    }
    // V fragment: B[k][n] = V[key=k][dim=n] from LDS-transposed tile
    v16h vf;
#pragma unroll
    for (int e = 0; e < 16; ++e)
      vf[e] = (hi == 0) ? (_Float16)lsV[wave][e * 16 + lo] : (_Float16)0.0f;
    oacc = __builtin_amdgcn_wmma_f32_16x16x32_f16(false, pf, false, vf,
                                                  (short)0, oacc, false, false);
  }
#pragma unroll
  for (int r = 0; r < 8; ++r) {
    int row = m0 + r + 8 * hi;
    O[(size_t)row * HD + head * DHD + lo] = oacc[r] / lrow[r];
  }
}

// ---------------- layernorm(x + res) * s + b, wave per row (b128 IO) ----------------
__global__ void k_ln_res(const float* __restrict__ x, const float* __restrict__ res,
                         const float* __restrict__ s, const float* __restrict__ b,
                         float* __restrict__ out) {
  int wave = threadIdx.x >> 5, lane = threadIdx.x & 31;
  int row = blockIdx.x * (blockDim.x >> 5) + wave;
  if (row >= NND) return;
  const float4 xv = *(const float4*)(x + (size_t)row * HD + lane * 4);
  const float4 rv = *(const float4*)(res + (size_t)row * HD + lane * 4);
  float v[4] = {xv.x + rv.x, xv.y + rv.y, xv.z + rv.z, xv.w + rv.w};
  float sum = v[0] + v[1] + v[2] + v[3];
  for (int d = 1; d < 32; d <<= 1) sum += __shfl_xor(sum, d, 32);
  float mean = sum / (float)HD;
  float var = 0.f;
#pragma unroll
  for (int i = 0; i < 4; ++i) { float t = v[i] - mean; var += t * t; }
  for (int d = 1; d < 32; d <<= 1) var += __shfl_xor(var, d, 32);
  var /= (float)HD;
  float inv = rsqrtf(var + 1e-5f);
  const float4 sv = *(const float4*)(s + lane * 4);
  const float4 bv = *(const float4*)(b + lane * 4);
  float4 ov;
  ov.x = (v[0] - mean) * inv * sv.x + bv.x;
  ov.y = (v[1] - mean) * inv * sv.y + bv.y;
  ov.z = (v[2] - mean) * inv * sv.z + bv.z;
  ov.w = (v[3] - mean) * inv * sv.w + bv.w;
  *(float4*)(out + (size_t)row * HD + lane * 4) = ov;
}

// ---------------- GATv2 edge passes ----------------
__device__ __forceinline__ unsigned fenc(float f) {
  unsigned u = __float_as_uint(f);
  return (u & 0x80000000u) ? ~u : (u | 0x80000000u);
}
__device__ __forceinline__ float fdec(unsigned u) {
  return __uint_as_float((u & 0x80000000u) ? (u & 0x7FFFFFFFu) : ~u);
}

__global__ void k_fill_u32(unsigned* p, unsigned v, int n) {
  int i = blockIdx.x * blockDim.x + threadIdx.x;
  if (i < n) p[i] = v;
}
__global__ void k_fill_f32(float* p, float v, int n) {
  int i = blockIdx.x * blockDim.x + threadIdx.x;
  if (i < n) p[i] = v;
}

__global__ void k_gat_logits(const int* __restrict__ ei, const float* __restrict__ xl,
                             const float* __restrict__ xr, const float* __restrict__ att,
                             float* __restrict__ logit, unsigned* __restrict__ mx) {
  int idx = blockIdx.x * blockDim.x + threadIdx.x;
  if (idx >= NEDGE * NHEADS) return;
  int e = idx / NHEADS, h = idx % NHEADS;
  int s = ei[e], d = ei[NEDGE + e];
  const float4* lp = (const float4*)(xl + (size_t)s * HD + h * DHD);
  const float4* rp = (const float4*)(xr + (size_t)d * HD + h * DHD);
  const float4* ap = (const float4*)(att + h * DHD);
  float acc = 0.f;
#pragma unroll
  for (int i = 0; i < 4; ++i) {
    float4 lv = lp[i], rv = rp[i], av = ap[i];
    float t0 = lv.x + rv.x; t0 = (t0 > 0.f) ? t0 : 0.2f * t0;
    float t1 = lv.y + rv.y; t1 = (t1 > 0.f) ? t1 : 0.2f * t1;
    float t2 = lv.z + rv.z; t2 = (t2 > 0.f) ? t2 : 0.2f * t2;
    float t3 = lv.w + rv.w; t3 = (t3 > 0.f) ? t3 : 0.2f * t3;
    acc += t0 * av.x + t1 * av.y + t2 * av.z + t3 * av.w;
  }
  logit[idx] = acc;
  atomicMax(&mx[d * NHEADS + h], fenc(acc));
}

__global__ void k_gat_exp(const int* __restrict__ ei, const unsigned* __restrict__ mx,
                          float* __restrict__ logit, float* __restrict__ denom) {
  int idx = blockIdx.x * blockDim.x + threadIdx.x;
  if (idx >= NEDGE * NHEADS) return;
  int e = idx / NHEADS, h = idx % NHEADS;
  int d = ei[NEDGE + e];
  float ex = expf(logit[idx] - fdec(mx[d * NHEADS + h]));
  logit[idx] = ex;
  atomicAdd(&denom[d * NHEADS + h], ex);
}

__global__ void k_gat_scatter(const int* __restrict__ ei, const float* __restrict__ ex,
                              const float* __restrict__ denom, const float* __restrict__ xl,
                              float* __restrict__ out) {
  int idx = blockIdx.x * blockDim.x + threadIdx.x;
  if (idx >= NEDGE * NHEADS) return;
  int e = idx / NHEADS, h = idx % NHEADS;
  int s = ei[e], d = ei[NEDGE + e];
  float alpha = ex[idx] / denom[d * NHEADS + h];
  const float4* lp = (const float4*)(xl + (size_t)s * HD + h * DHD);
  float* op = out + (size_t)d * HD + h * DHD;
#pragma unroll
  for (int i = 0; i < 4; ++i) {
    float4 lv = lp[i];
    atomicAdd(op + i * 4 + 0, alpha * lv.x);
    atomicAdd(op + i * 4 + 1, alpha * lv.y);
    atomicAdd(op + i * 4 + 2, alpha * lv.z);
    atomicAdd(op + i * 4 + 3, alpha * lv.w);
  }
}

__global__ void k_bias_add(const float* __restrict__ in, const float* __restrict__ bias,
                           float* __restrict__ out) {
  int idx = blockIdx.x * blockDim.x + threadIdx.x;
  if (idx >= NND * HD) return;
  out[idx] = in[idx] + bias[idx % HD];
}

// ---------------- classifier + time head -> out[N,4] ----------------
__global__ void k_head(const float* __restrict__ h, const float* __restrict__ cw,
                       const float* __restrict__ cb, const float* __restrict__ tw,
                       const float* __restrict__ tb, float* __restrict__ out) {
  int idx = blockIdx.x * blockDim.x + threadIdx.x;
  if (idx >= NND * 4) return;
  int n = idx >> 2, c = idx & 3;
  const float4* wp = (const float4*)((c < 3) ? (cw + c * HD) : tw);
  const float4* hp = (const float4*)(h + (size_t)n * HD);
  float acc = (c < 3) ? cb[c] : tb[0];
#pragma unroll 8
  for (int i = 0; i < HD / 4; ++i) {
    float4 hv = hp[i], wv = wp[i];
    acc += hv.x * wv.x + hv.y * wv.y + hv.z * wv.z + hv.w * wv.w;
  }
  out[idx] = acc;
}

extern "C" void kernel_launch(void* const* d_in, const int* in_sizes, int n_in,
                              void* d_out, int out_size, void* d_ws, size_t ws_size,
                              hipStream_t stream) {
  const float* x      = (const float*)d_in[0];
  const int*   ei     = (const int*)d_in[1];
  const float* emb_w  = (const float*)d_in[2];
  const float* emb_b  = (const float*)d_in[3];
  const float* ain_w  = (const float*)d_in[4];
  const float* ain_b  = (const float*)d_in[5];
  const float* aout_w = (const float*)d_in[6];
  const float* aout_b = (const float*)d_in[7];
  const float* glw    = (const float*)d_in[8];
  const float* glb    = (const float*)d_in[9];
  const float* grw    = (const float*)d_in[10];
  const float* grb    = (const float*)d_in[11];
  const float* gatt   = (const float*)d_in[12];
  const float* gbias  = (const float*)d_in[13];
  const float* einw   = (const float*)d_in[14];
  const float* einb   = (const float*)d_in[15];
  const float* eoutw  = (const float*)d_in[16];
  const float* eoutb  = (const float*)d_in[17];
  const float* ln1s   = (const float*)d_in[18];
  const float* ln1b   = (const float*)d_in[19];
  const float* w1     = (const float*)d_in[20];
  const float* b1     = (const float*)d_in[21];
  const float* w2     = (const float*)d_in[22];
  const float* b2     = (const float*)d_in[23];
  const float* ln2s   = (const float*)d_in[24];
  const float* ln2b   = (const float*)d_in[25];
  const float* clsw   = (const float*)d_in[26];
  const float* clsb   = (const float*)d_in[27];
  const float* timw   = (const float*)d_in[28];
  const float* timb   = (const float*)d_in[29];

  float* ws = (float*)d_ws;
  const size_t NH = (size_t)NND * HD;
  float*    h     = ws;                                     // [N,128]
  float*    t2    = ws + NH;                                // [N,128]
  float*    xl    = ws + 2 * NH;                            // [N,128]
  float*    xr    = ws + 3 * NH;                            // [N,128]
  unsigned* mx    = (unsigned*)(ws + 4 * NH);               // [N,8]
  float*    denom = ws + 4 * NH + (size_t)NND * NHEADS;     // [N,8]
  float*    t1    = ws + 4 * NH + 2 * (size_t)NND * NHEADS; // [N,2048] big scratch

  auto gemm = [&](const float* A, const float* W, const float* bias, float* C,
                  int M, int K, int Nout, int relu) {
    int blocks = (M / 16) * (Nout / 128);
    k_gemm_wmma<<<blocks, 256, 0, stream>>>(A, W, bias, C, M, K, Nout, relu);
  };

  // embed + positional encoding
  k_embed_pe<<<(NND * HD) / 256, 256, 0, stream>>>(x, emb_w, emb_b, h);

  // initial MHA (no residual): h = out_proj(attn(qkv(h)))
  gemm(h, ain_w, ain_b, t1, NND, HD, 3 * HD, 0);
  k_flash_attn<<<(NND / 16) * NHEADS / 8, 256, 0, stream>>>(t1, t2);
  gemm(t2, aout_w, aout_b, h, NND, HD, HD, 0);

  // GATv2 layers
  for (int l = 0; l < 2; ++l) {
    gemm(h, glw + (size_t)l * HD * HD, glb + l * HD, xl, NND, HD, HD, 0);
    gemm(h, grw + (size_t)l * HD * HD, grb + l * HD, xr, NND, HD, HD, 0);
    k_fill_u32<<<(NND * NHEADS) / 256, 256, 0, stream>>>(mx, 0u, NND * NHEADS);
    k_fill_f32<<<(NND * NHEADS) / 256, 256, 0, stream>>>(denom, 0.f, NND * NHEADS);
    k_fill_f32<<<(int)(NH / 256), 256, 0, stream>>>(t2, 0.f, (int)NH);
    int eth = NEDGE * NHEADS;
    k_gat_logits<<<eth / 256, 256, 0, stream>>>(ei, xl, xr,
                                                gatt + l * NHEADS * DHD, t1, mx);
    k_gat_exp<<<eth / 256, 256, 0, stream>>>(ei, mx, t1, denom);
    k_gat_scatter<<<eth / 256, 256, 0, stream>>>(ei, t1, denom, xl, t2);
    k_bias_add<<<(int)(NH / 256), 256, 0, stream>>>(t2, gbias + l * HD, h);
  }

  // transformer encoder layers (post-norm, ReLU FFN)
  for (int l = 0; l < 2; ++l) {
    gemm(h, einw + (size_t)l * 3 * HD * HD, einb + (size_t)l * 3 * HD, t1,
         NND, HD, 3 * HD, 0);
    k_flash_attn<<<(NND / 16) * NHEADS / 8, 256, 0, stream>>>(t1, t2);
    gemm(t2, eoutw + (size_t)l * HD * HD, eoutb + l * HD, xl, NND, HD, HD, 0);
    k_ln_res<<<NND / 8, 256, 0, stream>>>(h, xl, ln1s + l * HD, ln1b + l * HD, h);
    gemm(h, w1 + (size_t)l * FFD * HD, b1 + l * FFD, t1, NND, HD, FFD, 1);
    gemm(t1, w2 + (size_t)l * HD * FFD, b2 + l * HD, t2, NND, FFD, HD, 0);
    k_ln_res<<<NND / 8, 256, 0, stream>>>(h, t2, ln2s + l * HD, ln2b + l * HD, h);
  }

  // heads -> out[N,4]
  k_head<<<(NND * 4) / 256, 256, 0, stream>>>(h, clsw, clsb, timw, timb, (float*)d_out);
}
